// TransformerLanguageModel_66649302499618
// MI455X (gfx1250) — matrix operand order repaired and checked by reading, not compiled, not run on hardware
//
#include <hip/hip_runtime.h>

// ---------------------------------------------------------------------------
// Mini-GPT forward for MI455X (gfx1250, wave32, WMMA).
// All GEMMs run through v_wmma_f32_16x16x32_f16 with f16 activations/weights
// and f32 accumulation. Weights are pre-transposed to [N][K] f16 once per
// launch so every LDS fragment load is a contiguous 16B/32B ds_load_b128.
// K-tiles are streamed into double-buffered LDS with the CDNA5 async path
// (global_load_async_to_lds_b128 + s_wait_asynccnt), overlapping global
// traffic with WMMA compute.
// ---------------------------------------------------------------------------

typedef _Float16 f16;
typedef __attribute__((ext_vector_type(16))) _Float16 v16h;
typedef __attribute__((ext_vector_type(8)))  float    v8f;

constexpr int B_   = 128;
constexpr int T_   = 128;
constexpr int C_   = 256;
constexpr int H_   = 4;
constexpr int L_   = 6;
constexpr int V_   = 10000;
constexpr int HS_  = 64;
constexpr int DFF_ = 1024;
constexpr int M_   = B_ * T_;   // 16384 token rows

union Frag {
  v16h  h;
  float4 f4[2];
};

__device__ inline v8f wmma16(const Frag& a, const Frag& b, v8f c) {
  // D = A(16x32 f16) * B(32x16 f16) + C(16x16 f32)
  return __builtin_amdgcn_wmma_f32_16x16x32_f16(
      /*neg_a=*/false, a.h, /*neg_b=*/false, b.h,
      /*c_mod=*/(short)0, c, /*reuse_a=*/false, /*reuse_b=*/false);
}

__device__ inline v8f v8f_zero() {
  v8f z = {0.f, 0.f, 0.f, 0.f, 0.f, 0.f, 0.f, 0.f};
  return z;
}

// Async 16B copy global -> LDS, tracked by ASYNCcnt (no VGPR data path).
// vdst VGPR carries the LDS byte address (low 32 bits of the generic shared
// pointer == LDS offset per the aperture rules); vaddr is the 64-bit VA.
__device__ inline void async_copy_b128(const void* gptr, void* lptr) {
  unsigned lds = (unsigned)(size_t)lptr;
  unsigned long long ga = (unsigned long long)(size_t)gptr;
  asm volatile("global_load_async_to_lds_b128 %0, %1, off"
               :: "v"(lds), "v"(ga) : "memory");
}

__device__ inline void wait_async0() {
  asm volatile("s_wait_asynccnt 0x0" ::: "memory");
}

// ---------------------------------------------------------------------------
// Weight conversion: f32 [K,N] row-major  ->  f16 [N,K] row-major (transposed)
// ---------------------------------------------------------------------------
__global__ void convT_kernel(const float* __restrict__ src, f16* __restrict__ dst,
                             int Kd, int Nd) {
  size_t i = (size_t)blockIdx.x * 256 + threadIdx.x;
  if (i >= (size_t)Kd * Nd) return;
  int k = (int)(i / Nd);
  int n = (int)(i % Nd);
  dst[(size_t)n * Kd + k] = (f16)src[i];
}

// ---------------------------------------------------------------------------
// Embedding: x[row, c] = tok_emb[idx[row], c] + pos_emb[row % T, c]
// ---------------------------------------------------------------------------
__global__ void embed_kernel(const int* __restrict__ idx,
                             const float* __restrict__ tok,
                             const float* __restrict__ pos,
                             float* __restrict__ X) {
  int row = blockIdx.x;
  int c   = threadIdx.x;
  int t   = row % T_;
  int tk  = idx[row];
  X[(size_t)row * C_ + c] = tok[(size_t)tk * C_ + c] + pos[(size_t)t * C_ + c];
}

// ---------------------------------------------------------------------------
// LayerNorm (C == blockDim == 256), f32 in -> f16 out
// ---------------------------------------------------------------------------
__global__ __launch_bounds__(256) void layernorm_kernel(
    const float* __restrict__ X, const float* __restrict__ G,
    const float* __restrict__ Bv, f16* __restrict__ O) {
  int row = blockIdx.x, tid = threadIdx.x;
  int lane = tid & 31, wave = tid >> 5;
  float v  = X[(size_t)row * C_ + tid];
  float s  = v, s2 = v * v;
  for (int m = 16; m >= 1; m >>= 1) {
    s  += __shfl_xor(s, m, 32);
    s2 += __shfl_xor(s2, m, 32);
  }
  __shared__ float ws1[8], ws2[8];
  if (lane == 0) { ws1[wave] = s; ws2[wave] = s2; }
  __syncthreads();
  if (wave == 0) {
    float a  = (lane < 8) ? ws1[lane] : 0.f;
    float b2 = (lane < 8) ? ws2[lane] : 0.f;
    for (int m = 4; m >= 1; m >>= 1) {
      a  += __shfl_xor(a, m, 8);
      b2 += __shfl_xor(b2, m, 8);
    }
    if (lane == 0) { ws1[0] = a; ws2[0] = b2; }
  }
  __syncthreads();
  float mean = ws1[0] * (1.0f / C_);
  float var  = ws2[0] * (1.0f / C_) - mean * mean;
  float rstd = rsqrtf(var + 1e-5f);
  O[(size_t)row * C_ + tid] = (f16)((v - mean) * rstd * G[tid] + Bv[tid]);
}

// ---------------------------------------------------------------------------
// WMMA GEMM: C[M,N] = A[M,K] * Bt[N,K]^T (+bias) (+relu) (+residual)
// Block = 256 threads = 8 waves (2x4), block tile 128x128, K step 32.
// Each wave: 64x32 tile = 4x2 accumulators, 8 v_wmma per K-step.
// Double-buffered LDS fed by global_load_async_to_lds_b128 (ASYNCcnt).
// OUTMODE: 0 = f16 store, 1 = f16 store + relu, 2 = f32 "+=", 3 = f32 store
// ---------------------------------------------------------------------------
template <int OUTMODE>
__global__ __launch_bounds__(256) void gemm_kernel(
    const f16* __restrict__ A, const f16* __restrict__ Bt,
    const float* __restrict__ bias, void* __restrict__ Out,
    int M, int N, int K) {
  constexpr int BM = 128, BN = 128, BK = 32;
  constexpr int LDT = BK + 8;              // 40 halves: 80B row, 16B aligned
  __shared__ f16 Als[2][BM * LDT];
  __shared__ f16 Bls[2][BN * LDT];

  const int m0 = blockIdx.x * BM, n0 = blockIdx.y * BN;
  const int tid  = threadIdx.x;
  const int lane = tid & 31, wave = tid >> 5;
  const int half = lane >> 4, lr = lane & 15;
  const int wm = (wave & 1) * 64;          // wave row base in tile
  const int wn = (wave >> 1) * 32;         // wave col base in tile

  // Per-thread async staging: 2 x 16B chunks per matrix per tile.
  const int r0   = tid >> 2;               // chunk row for i=0 (0..63)
  const int off0 = (tid & 3) * 8;          // chunk K-offset in halves
  auto issue_tile = [&](int kt, int buf) {
    int k0 = kt * BK;
    for (int i = 0; i < 2; ++i) {
      int r = r0 + i * 64;
      if (m0 + r < M)
        async_copy_b128(A + (size_t)(m0 + r) * K + k0 + off0,
                        &Als[buf][r * LDT + off0]);
      if (n0 + r < N)
        async_copy_b128(Bt + (size_t)(n0 + r) * K + k0 + off0,
                        &Bls[buf][r * LDT + off0]);
    }
  };

  v8f acc[4][2];
  for (int mm = 0; mm < 4; ++mm)
    for (int nn = 0; nn < 2; ++nn) acc[mm][nn] = v8f_zero();

  const int NT = K / BK;
  issue_tile(0, 0);

  for (int kt = 0; kt < NT; ++kt) {
    int buf = kt & 1;
    wait_async0();            // own tile-kt transfers complete
    __syncthreads();          // everyone's transfers visible
    if (kt + 1 < NT) issue_tile(kt + 1, buf ^ 1);   // overlap with compute

    // A-frag (16x32): lane=row, halves [half*8, +8) and [half*8+16, +8)
    Frag af[4];
    for (int mm = 0; mm < 4; ++mm) {
      const f16* p = &Als[buf][(wm + mm * 16 + lr) * LDT + half * 8];
      af[mm].f4[0] = *(const float4*)p;
      af[mm].f4[1] = *(const float4*)(p + 16);
    }
    // B-frag (32x16): lane=col, halves [half*8, +8) and [half*8+16, +8)
    Frag bf[2];
    for (int nn = 0; nn < 2; ++nn) {
      const f16* p = &Bls[buf][(wn + nn * 16 + lr) * LDT + half * 8];
      bf[nn].f4[0] = *(const float4*)p;
      bf[nn].f4[1] = *(const float4*)(p + 16);
    }
    for (int mm = 0; mm < 4; ++mm)
      for (int nn = 0; nn < 2; ++nn)
        acc[mm][nn] = wmma16(af[mm], bf[nn], acc[mm][nn]);
    __syncthreads();          // done reading buf before tile kt+2 overwrites
  }

  // Epilogue (C layout: vgpr j -> row j + half*8, col = lr)
  for (int mm = 0; mm < 4; ++mm)
    for (int nn = 0; nn < 2; ++nn)
      for (int j = 0; j < 8; ++j) {
        int gr = m0 + wm + mm * 16 + j + half * 8;
        int gc = n0 + wn + nn * 16 + lr;
        if (gr < M && gc < N) {
          float v = acc[mm][nn][j];
          if (bias) v += bias[gc];
          if constexpr (OUTMODE == 1) v = fmaxf(v, 0.f);
          size_t o = (size_t)gr * N + gc;
          if constexpr (OUTMODE <= 1)
            ((f16*)Out)[o] = (f16)v;
          else if constexpr (OUTMODE == 2)
            ((float*)Out)[o] += v;       // residual accumulate into x
          else
            ((float*)Out)[o] = v;
        }
      }
}

// ---------------------------------------------------------------------------
// Causal attention: one block (256 thr, 8 waves) per (b,h).
// S = Q K^T via WMMA, softmax in C-layout registers, O = P V via WMMA.
// LDS: Vt[64][136] + union{K[128][72], P[128][136]} = 52224 B.
// K tile is staged with the async-to-LDS path.
// ---------------------------------------------------------------------------
__global__ __launch_bounds__(256) void attn_kernel(
    const f16* __restrict__ Qm, const f16* __restrict__ Km,
    const f16* __restrict__ Vm, f16* __restrict__ Om) {
  constexpr int KLD = 72;    // 144B rows (16B aligned, bank-rotating)
  constexpr int VLD = 136;   // 272B rows
  constexpr int PLD = 136;
  constexpr int KP_HALVES = (T_ * KLD > 128 * PLD) ? T_ * KLD : 128 * PLD;
  __shared__ f16 Vt[HS_ * VLD];
  __shared__ f16 KP[KP_HALVES];
  f16* Kls = KP;
  f16* Pls = KP;   // reused after barrier

  const int bh = blockIdx.x;
  const int b = bh / H_, h = bh % H_;
  const int tid = threadIdx.x, lane = tid & 31, wave = tid >> 5;
  const int half = lane >> 4, lr = lane & 15;
  const size_t rowBase = (size_t)b * T_ * C_ + (size_t)h * HS_;

  // Stage K[key][d] asynchronously (1024 chunks of 8 halves)
  for (int i = 0; i < 4; ++i) {
    int c = tid + i * 256;
    int row = c >> 3, off = (c & 7) * 8;
    async_copy_b128(Km + rowBase + (size_t)row * C_ + off,
                    &Kls[row * KLD + off]);
  }
  // Stage Vt[d][key] (transposed so PV B-frags are K-contiguous)
  for (int i = 0; i < 32; ++i) {
    int c = tid + i * 256;           // 8192 scalars
    int d = c & 63, key = c >> 6;
    Vt[d * VLD + key] = Vm[rowBase + (size_t)key * C_ + d];
  }
  wait_async0();
  __syncthreads();

  // Q A-frags for this wave's 16 query rows, straight from global
  Frag qf[2];
  {
    const f16* qp = Qm + rowBase + (size_t)(wave * 16 + lr) * C_;
    for (int kc = 0; kc < 2; ++kc) {
      const f16* p = qp + kc * 32 + half * 8;
      qf[kc].f4[0] = *(const float4*)p;
      qf[kc].f4[1] = *(const float4*)(p + 16);
    }
  }

  // Scores: 8 key-blocks of 16, K-dim 64 = 2 wmma each
  v8f sc[8];
  for (int nb = 0; nb < 8; ++nb) {
    v8f a = v8f_zero();
    for (int kc = 0; kc < 2; ++kc) {
      Frag kf;
      const f16* p = &Kls[(nb * 16 + lr) * KLD + kc * 32 + half * 8];
      kf.f4[0] = *(const float4*)p;
      kf.f4[1] = *(const float4*)(p + 16);
      a = wmma16(qf[kc], kf, a);
    }
    sc[nb] = a;
  }
  __syncthreads();   // all waves done with Kls; region becomes Pls

  // Causal softmax. Row r = wave*16 + j + half*8 lives in vgpr j, 16 lanes of
  // one half-wave hold its 8x16 = 128 columns.
  const float scale = 0.0625f;   // C^-0.5 = 1/16 (note: C, not head_size)
  for (int j = 0; j < 8; ++j) {
    int qrow = wave * 16 + j + half * 8;
    float vals[8];
    float mx = -3.0e38f;
    for (int nb = 0; nb < 8; ++nb) {
      int col = nb * 16 + lr;
      float s = (col <= qrow) ? (float)sc[nb][j] * scale : -3.0e38f;
      vals[nb] = s;
      mx = fmaxf(mx, s);
    }
    for (int m = 8; m >= 1; m >>= 1) mx = fmaxf(mx, __shfl_xor(mx, m, 16));
    float sum = 0.f;
    for (int nb = 0; nb < 8; ++nb) {
      float e = __expf(vals[nb] - mx);
      vals[nb] = e;
      sum += e;
    }
    for (int m = 8; m >= 1; m >>= 1) sum += __shfl_xor(sum, m, 16);
    float inv = 1.0f / sum;
    for (int nb = 0; nb < 8; ++nb)
      Pls[(size_t)qrow * PLD + nb * 16 + lr] = (f16)(vals[nb] * inv);
  }
  __syncthreads();

  // O = P[16,128] * V[128,64] per wave: 4 d-blocks x 4 K-chunks of 32
  for (int db = 0; db < 4; ++db) {
    v8f o = v8f_zero();
    for (int kc = 0; kc < 4; ++kc) {
      Frag pf, vf;
      const f16* pp = &Pls[(size_t)(wave * 16 + lr) * PLD + kc * 32 + half * 8];
      pf.f4[0] = *(const float4*)pp;
      pf.f4[1] = *(const float4*)(pp + 16);
      const f16* vp = &Vt[(db * 16 + lr) * VLD + kc * 32 + half * 8];
      vf.f4[0] = *(const float4*)vp;
      vf.f4[1] = *(const float4*)(vp + 16);
      o = wmma16(pf, vf, o);
    }
    for (int j = 0; j < 8; ++j) {
      int qrow = wave * 16 + j + half * 8;
      Om[rowBase + (size_t)qrow * C_ + db * 16 + lr] = (f16)o[j];
    }
  }
}

// ---------------------------------------------------------------------------
// Host-side orchestration
// ---------------------------------------------------------------------------
extern "C" void kernel_launch(void* const* d_in, const int* in_sizes, int n_in,
                              void* d_out, int out_size, void* d_ws, size_t ws_size,
                              hipStream_t stream) {
  (void)in_sizes; (void)n_in; (void)out_size; (void)ws_size;

  const int*   idx     = (const int*)  d_in[0];
  const float* tok_emb = (const float*)d_in[1];
  const float* pos_emb = (const float*)d_in[2];
  const float* ln1_g   = (const float*)d_in[3];
  const float* ln1_b   = (const float*)d_in[4];
  const float* wq      = (const float*)d_in[5];
  const float* wk      = (const float*)d_in[6];
  const float* wv      = (const float*)d_in[7];
  const float* proj_w  = (const float*)d_in[8];
  const float* proj_b  = (const float*)d_in[9];
  const float* ln2_g   = (const float*)d_in[10];
  const float* ln2_b   = (const float*)d_in[11];
  const float* w1      = (const float*)d_in[12];
  const float* b1      = (const float*)d_in[13];
  const float* w2      = (const float*)d_in[14];
  const float* b2      = (const float*)d_in[15];
  const float* lnf_g   = (const float*)d_in[16];
  const float* lnf_b   = (const float*)d_in[17];
  const float* lm_w    = (const float*)d_in[18];
  const float* lm_b    = (const float*)d_in[19];

  // Workspace carve-up (256B aligned)
  char* wsb = (char*)d_ws;
  size_t off = 0;
  auto bump = [&](size_t bytes) -> void* {
    void* p = wsb + off;
    off = (off + bytes + 255) & ~(size_t)255;
    return p;
  };
  float* xf32  = (float*)bump((size_t)M_ * C_ * 4);
  f16*   h16   = (f16*)  bump((size_t)M_ * C_ * 2);
  f16*   q16   = (f16*)  bump((size_t)M_ * C_ * 2);
  f16*   k16   = (f16*)  bump((size_t)M_ * C_ * 2);
  f16*   v16   = (f16*)  bump((size_t)M_ * C_ * 2);
  f16*   att16 = (f16*)  bump((size_t)M_ * C_ * 2);
  f16*   t116  = (f16*)  bump((size_t)M_ * DFF_ * 2);
  f16*   wqT   = (f16*)  bump((size_t)L_ * C_ * C_ * 2);
  f16*   wkT   = (f16*)  bump((size_t)L_ * C_ * C_ * 2);
  f16*   wvT   = (f16*)  bump((size_t)L_ * C_ * C_ * 2);
  f16*   projT = (f16*)  bump((size_t)L_ * C_ * C_ * 2);
  f16*   w1T   = (f16*)  bump((size_t)L_ * C_ * DFF_ * 2);
  f16*   w2T   = (f16*)  bump((size_t)L_ * DFF_ * C_ * 2);
  f16*   lmT   = (f16*)  bump((size_t)C_ * V_ * 2);

  dim3 blk(256);
  auto cdiv = [](int a, int b) { return (a + b - 1) / b; };

  // Weight conversion to transposed f16 (deterministic, every launch)
  for (int l = 0; l < L_; ++l) {
    size_t oc = (size_t)l * C_ * C_;
    convT_kernel<<<cdiv(C_ * C_, 256), blk, 0, stream>>>(wq + oc, wqT + oc, C_, C_);
    convT_kernel<<<cdiv(C_ * C_, 256), blk, 0, stream>>>(wk + oc, wkT + oc, C_, C_);
    convT_kernel<<<cdiv(C_ * C_, 256), blk, 0, stream>>>(wv + oc, wvT + oc, C_, C_);
    convT_kernel<<<cdiv(C_ * C_, 256), blk, 0, stream>>>(proj_w + oc, projT + oc, C_, C_);
    size_t o1 = (size_t)l * C_ * DFF_;
    convT_kernel<<<cdiv(C_ * DFF_, 256), blk, 0, stream>>>(w1 + o1, w1T + o1, C_, DFF_);
    convT_kernel<<<cdiv(DFF_ * C_, 256), blk, 0, stream>>>(w2 + o1, w2T + o1, DFF_, C_);
  }
  convT_kernel<<<cdiv(C_ * V_, 256), blk, 0, stream>>>(lm_w, lmT, C_, V_);

  // Embedding
  embed_kernel<<<M_, blk, 0, stream>>>(idx, tok_emb, pos_emb, xf32);

  dim3 gC(M_ / 128, cdiv(C_, 128));      // N = 256
  dim3 gF(M_ / 128, cdiv(DFF_, 128));    // N = 1024
  dim3 gV(M_ / 128, cdiv(V_, 128));      // N = 10000 (partial last tile)

  for (int l = 0; l < L_; ++l) {
    size_t oc = (size_t)l * C_ * C_;
    size_t o1 = (size_t)l * C_ * DFF_;
    layernorm_kernel<<<M_, blk, 0, stream>>>(xf32, ln1_g + l * C_, ln1_b + l * C_, h16);
    gemm_kernel<0><<<gC, blk, 0, stream>>>(h16, wqT + oc, nullptr, q16, M_, C_, C_);
    gemm_kernel<0><<<gC, blk, 0, stream>>>(h16, wkT + oc, nullptr, k16, M_, C_, C_);
    gemm_kernel<0><<<gC, blk, 0, stream>>>(h16, wvT + oc, nullptr, v16, M_, C_, C_);
    attn_kernel<<<B_ * H_, blk, 0, stream>>>(q16, k16, v16, att16);
    gemm_kernel<2><<<gC, blk, 0, stream>>>(att16, projT + oc, proj_b + l * C_, xf32, M_, C_, C_);
    layernorm_kernel<<<M_, blk, 0, stream>>>(xf32, ln2_g + l * C_, ln2_b + l * C_, h16);
    gemm_kernel<1><<<gF, blk, 0, stream>>>(h16, w1T + o1, b1 + l * DFF_, t116, M_, DFF_, C_);
    gemm_kernel<2><<<gC, blk, 0, stream>>>(t116, w2T + o1, b2 + l * C_, xf32, M_, C_, DFF_);
  }

  layernorm_kernel<<<M_, blk, 0, stream>>>(xf32, lnf_g, lnf_b, h16);
  gemm_kernel<3><<<gV, blk, 0, stream>>>(h16, lmT, lm_b, d_out, M_, V_, C_);
}
